// ClusterMemory_30408368456272
// MI455X (gfx1250) — matrix operand order repaired and checked by reading, not compiled, not run on hardware
//
#include <hip/hip_runtime.h>
#include <hip/hip_bf16.h>

typedef __attribute__((ext_vector_type(16))) _Float16 v16h;
typedef __attribute__((ext_vector_type(8)))  _Float16 v8h;
typedef __attribute__((ext_vector_type(8)))  float    v8f;

#define BATCH 256
#define DIM 256
#define NROWS 400000
#define NCHUNK 80                 // feature rows per workgroup (80 | 400000)
#define NWG (NROWS / NCHUNK)      // 5000
#define INV_TEMP 20.0f            // 1 / 0.05
#define SHIFT 20.0f               // max possible logit (|dot| <= 1)
#define MOM 0.2f
#define LDSTR 264                 // padded LDS row stride in f16 (bank-friendly)

// ---------------------------------------------------------------- prep:
// x = inputs / ||inputs||  -> xf (f32, for update/target kernels) and xh (f16, WMMA A)
__global__ __launch_bounds__(256) void prep_kernel(const float* __restrict__ inputs,
                                                   float* __restrict__ xf,
                                                   _Float16* __restrict__ xh) {
  __shared__ float red[256];
  const int i = blockIdx.x, d = threadIdx.x;
  float v = inputs[i * DIM + d];
  red[d] = v * v;
  __syncthreads();
  for (int s = 128; s > 0; s >>= 1) { if (d < s) red[d] += red[d + s]; __syncthreads(); }
  float x = v / sqrtf(red[0]);
  xf[i * DIM + d] = x;
  xh[i * DIM + d] = (_Float16)x;
}

// ---------------------------------------------------------------- fused GEMM + exp-sum + copy:
// For an 80-row chunk of features: stage rows in LDS as f16, compute
// logits = (x @ chunk^T) * 20 via v_wmma_f32_16x16x32_f16, accumulate
// sum_n exp(logit - 20) per batch row into part[wg][m], and stream the
// (unchanged) feature rows to the output with non-temporal stores.
__global__ __launch_bounds__(256) void gemm_softmax_kernel(const _Float16* __restrict__ xh,
                                                           const float* __restrict__ feat,
                                                           float* __restrict__ part,
                                                           float* __restrict__ outF) {
  __shared__ _Float16 Bsh[NCHUNK * LDSTR];
  __shared__ float rowsum[BATCH];
  const int tid = threadIdx.x;
  const long n0 = (long)blockIdx.x * NCHUNK;

  // stage 80x256 f32 -> f16 LDS, and copy-through to output (4-byte aligned dst)
#pragma unroll
  for (int it = 0; it < (NCHUNK * DIM / 4) / 256; ++it) {   // 20 iterations
    int idx = tid + it * 256;
    int r = idx >> 6;              // 64 float4 per row
    int d4 = (idx & 63) << 2;
    float4 v = *(const float4*)(feat + (n0 + r) * DIM + d4);
    _Float16* p = &Bsh[r * LDSTR + d4];
    p[0] = (_Float16)v.x; p[1] = (_Float16)v.y; p[2] = (_Float16)v.z; p[3] = (_Float16)v.w;
    float* dp = outF + (n0 + r) * DIM + d4;
    __builtin_nontemporal_store(v.x, dp + 0);
    __builtin_nontemporal_store(v.y, dp + 1);
    __builtin_nontemporal_store(v.z, dp + 2);
    __builtin_nontemporal_store(v.w, dp + 3);
  }
  __syncthreads();

  const int wave  = tid >> 5;
  const int lane  = tid & 31;
  const int kh    = lane >> 4;     // lane-half selects K sub-block (ISA A/B layouts)
  const int l16   = lane & 15;

  for (int t = 0; t < 2; ++t) {
    const int mt = wave * 2 + t;                    // 16-row M tile index (0..15)
    // preload all A fragments (K = 0..255) for this M tile: per-lane two 16B chunks per K-step
    const _Float16* ap = xh + (mt * 16 + l16) * DIM + kh * 8;
    v16h afr[8];
#pragma unroll
    for (int k0 = 0; k0 < 8; ++k0) {
      v8h lo = *(const v8h*)(ap + k0 * 32);         // K = 32k0 + kh*8 + [0..7]
      v8h hi = *(const v8h*)(ap + k0 * 32 + 16);    // K = 32k0 + 16 + kh*8 + [0..7]
      afr[k0] = __builtin_shufflevector(lo, hi, 0,1,2,3,4,5,6,7,8,9,10,11,12,13,14,15);
    }
    float acc[8] = {0.f,0.f,0.f,0.f,0.f,0.f,0.f,0.f};
    for (int nt = 0; nt < NCHUNK / 16; ++nt) {      // 5 N tiles
      v8f c = {};
      const _Float16* bp = &Bsh[(nt * 16 + l16) * LDSTR + kh * 8];
#pragma unroll
      for (int k0 = 0; k0 < 8; ++k0) {
        v8h blo = *(const v8h*)(bp + k0 * 32);      // K = 32k0 + kh*8 + [0..7]
        v8h bhi = *(const v8h*)(bp + k0 * 32 + 16); // K = 32k0 + 16 + kh*8 + [0..7]
        v16h b = __builtin_shufflevector(blo, bhi, 0,1,2,3,4,5,6,7,8,9,10,11,12,13,14,15);
        c = __builtin_amdgcn_wmma_f32_16x16x32_f16(false, afr[k0], false, b,
                                                   (short)0, c, false, false);
      }
      // per-lane accumulation only (VALU); cross-lane reduce deferred to after N loop
#pragma unroll
      for (int j = 0; j < 8; ++j)
        acc[j] += __expf(fmaf(c[j], INV_TEMP, -SHIFT));
    }
    // single cross-lane row-sum per M tile (4 xor steps over the 16 N lanes of each half)
#pragma unroll
    for (int j = 0; j < 8; ++j) {
      float e = acc[j];
      e += __shfl_xor(e, 1, 32);
      e += __shfl_xor(e, 2, 32);
      e += __shfl_xor(e, 4, 32);
      e += __shfl_xor(e, 8, 32);
      acc[j] = e;
    }
    if (l16 == 0) {
      int mrow = mt * 16 + kh * 8;                  // C layout: M = j + 8*(lane>>4)
#pragma unroll
      for (int j = 0; j < 8; ++j) rowsum[mrow + j] = acc[j];
    }
  }
  __syncthreads();
  part[(long)blockIdx.x * BATCH + tid] = rowsum[tid];   // coalesced partial write
}

// ---------------------------------------------------------------- deterministic reduction of partials
__global__ __launch_bounds__(256) void reduce_kernel(const float* __restrict__ part,
                                                     float* __restrict__ sumexp) {
  __shared__ float red[256];
  const int m = blockIdx.x;
  float s = 0.f;
  for (int w = threadIdx.x; w < NWG; w += 256) s += part[(long)w * BATCH + m];
  red[threadIdx.x] = s;
  __syncthreads();
  for (int st = 128; st > 0; st >>= 1) { if (threadIdx.x < st) red[threadIdx.x] += red[threadIdx.x + st]; __syncthreads(); }
  if (threadIdx.x == 0) sumexp[m] = red[0];
}

// ---------------------------------------------------------------- target logits (exact f32 dots)
__global__ __launch_bounds__(256) void tlogit_kernel(const float* __restrict__ xf,
                                                     const float* __restrict__ feat,
                                                     const int* __restrict__ targets,
                                                     float* __restrict__ tlog) {
  __shared__ float red[256];
  const int i = blockIdx.x, d = threadIdx.x;
  const int t = targets[i];
  red[d] = xf[i * DIM + d] * feat[(long)t * DIM + d];
  __syncthreads();
  for (int s = 128; s > 0; s >>= 1) { if (d < s) red[d] += red[d + s]; __syncthreads(); }
  if (d == 0) tlog[i] = red[0] * INV_TEMP;
}

// ---------------------------------------------------------------- loss = mean(logsumexp - tlogit)
__global__ __launch_bounds__(256) void loss_kernel(const float* __restrict__ sumexp,
                                                   const float* __restrict__ tlog,
                                                   float* __restrict__ out) {
  __shared__ float red[256];
  const int i = threadIdx.x;
  red[i] = logf(sumexp[i]) + SHIFT - tlog[i];
  __syncthreads();
  for (int s = 128; s > 0; s >>= 1) { if (i < s) red[i] += red[i + s]; __syncthreads(); }
  if (i == 0) out[0] = red[0] / (float)BATCH;
}

// ---------------------------------------------------------------- ordered memory-bank update
// Block i handles batch sample i; only the first occurrence of a target runs the
// whole (ordered) chain for that row, so duplicate targets stay sequential.
__global__ __launch_bounds__(256) void update_kernel(const float* __restrict__ feat,
                                                     const int* __restrict__ targets,
                                                     const float* __restrict__ xf,
                                                     float* __restrict__ outF) {
  __shared__ int tg[BATCH];
  __shared__ float red[256];
  const int tid = threadIdx.x;
  const int i = blockIdx.x;
  tg[tid] = targets[tid];
  __syncthreads();
  const int y = tg[i];
  for (int j = 0; j < i; ++j)
    if (tg[j] == y) return;                 // uniform exit: an earlier block owns this chain
  float f = feat[(long)y * DIM + tid];      // start from original row
  for (int j = i; j < BATCH; ++j) {
    if (tg[j] == y) {                       // uniform per block
      f = MOM * f + (1.0f - MOM) * xf[j * DIM + tid];
      red[tid] = f * f;
      __syncthreads();
      for (int s = 128; s > 0; s >>= 1) { if (tid < s) red[tid] += red[tid + s]; __syncthreads(); }
      f /= sqrtf(red[0]);
      __syncthreads();
    }
  }
  outF[(long)y * DIM + tid] = f;            // overwrites the copied-through row
}

// ----------------------------------------------------------------
extern "C" void kernel_launch(void* const* d_in, const int* in_sizes, int n_in,
                              void* d_out, int out_size, void* d_ws, size_t ws_size,
                              hipStream_t stream) {
  (void)in_sizes; (void)n_in; (void)out_size; (void)ws_size;
  const float* inputs  = (const float*)d_in[0];
  const int*   targets = (const int*)d_in[1];
  const float* feat    = (const float*)d_in[2];
  float* out  = (float*)d_out;
  float* outF = out + 1;                     // new_features, 4-byte aligned only

  char* ws = (char*)d_ws;
  float*    xf     = (float*)ws;                       // 256 KB
  _Float16* xh     = (_Float16*)(ws + 262144);         // 128 KB
  float*    part   = (float*)(ws + 393216);            // 5000*256*4 = 5.12 MB
  float*    sumexp = (float*)(ws + 5513216);           // 1 KB
  float*    tlog   = (float*)(ws + 5514240);           // 1 KB

  prep_kernel<<<BATCH, 256, 0, stream>>>(inputs, xf, xh);
  gemm_softmax_kernel<<<NWG, 256, 0, stream>>>(xh, feat, part, outF);
  reduce_kernel<<<BATCH, 256, 0, stream>>>(part, sumexp);
  tlogit_kernel<<<BATCH, 256, 0, stream>>>(xf, feat, targets, tlog);
  loss_kernel<<<1, 256, 0, stream>>>(sumexp, tlog, out);
  update_kernel<<<BATCH, 256, 0, stream>>>(feat, targets, xf, outF);
}
